// QLoRALinear_19756849561799
// MI455X (gfx1250) — matrix operand order repaired and checked by reading, not compiled, not run on hardware
//
#include <hip/hip_runtime.h>
#include <hip/hip_bf16.h>

// ---------------------------------------------------------------------------
// QLoRA linear for MI455X (gfx1250, wave32, WMMA).
//   out = x @ dequant(qweight,scales)^T  +  4 * (x @ A^T) @ B^T
// M=8192, N=4096, K=4096. Compute-bound (AI ~900 FLOP/B) -> f16 WMMA with
// f32 accumulation (v_wmma_f32_16x16x32_f16). Global loads for slab k+1 are
// register-staged and issued under slab k's WMMA phase to hide VMEM latency.
// ---------------------------------------------------------------------------

#define TOKENS 8192
#define IN_F   4096
#define OUT_F  4096
#define RANK   8
#define BM     128
#define BN     128
#define BK     64          // == NF4 block size -> one scalar scale per K slab
#define NSLAB  (IN_F / BK)
#define LDSX   72          // halves per LDS row (144 B, 16B-aligned, conflict-free)

typedef _Float16 v16h __attribute__((ext_vector_type(16)));
typedef _Float16 v8h  __attribute__((ext_vector_type(8)));
typedef _Float16 v4h  __attribute__((ext_vector_type(4)));
typedef float    v8f  __attribute__((ext_vector_type(8)));
typedef float    v4f  __attribute__((ext_vector_type(4)));
typedef int      v4i  __attribute__((ext_vector_type(4)));

__constant__ float NF4_TAB[16] = {
    -1.0f, -0.6961928009986877f, -0.5250730514526367f, -0.39491748809814453f,
    -0.28444138169288635f, -0.18477343022823334f, -0.09105003625154495f, 0.0f,
    0.07958029955625534f, 0.16093020141124725f, 0.24611230194568634f,
    0.33791524171829224f, 0.44070982933044434f, 0.5626170039176941f,
    0.7239560484886169f, 1.0f};

// ---------------------------------------------------------------------------
// Kernel 1: t[m][r] = 4 * sum_k x[m][k] * A[r][k]   (8192 x 8, into d_ws)
// ---------------------------------------------------------------------------
__global__ __launch_bounds__(256) void lora_xa_kernel(
    const float* __restrict__ x, const float* __restrict__ A,
    float* __restrict__ t) {
  const int m = blockIdx.x * 256 + threadIdx.x;
  float acc[RANK];
#pragma unroll
  for (int r = 0; r < RANK; ++r) acc[r] = 0.0f;

  const float* xrow = x + (size_t)m * IN_F;
  for (int k = 0; k < IN_F; k += 4) {
    v4f xv = *(const v4f*)(xrow + k);
#pragma unroll
    for (int r = 0; r < RANK; ++r) {
      v4f av = *(const v4f*)(A + (size_t)r * IN_F + k);
      acc[r] += xv.x * av.x + xv.y * av.y + xv.z * av.z + xv.w * av.w;
    }
  }
#pragma unroll
  for (int r = 0; r < RANK; ++r) t[(size_t)m * RANK + r] = 4.0f * acc[r];
}

// ---------------------------------------------------------------------------
// Kernel 2: WMMA GEMM with fused NF4 dequant + LoRA epilogue.
// Block: 256 threads = 8 waves, 4(M) x 2(N) wave grid.
// Wave tile: 32x64 -> 2 m-frags x 4 n-frags, 8x v8f accumulators.
// Pipeline: slab k+1 globals issued under slab k's WMMA phase.
// ---------------------------------------------------------------------------
__global__ __launch_bounds__(256) void qlora_wmma_kernel(
    const float* __restrict__ x, const int* __restrict__ qw,
    const float* __restrict__ scales, const float* __restrict__ loraB,
    const float* __restrict__ t, float* __restrict__ out) {

  __shared__ _Float16 xs[BM * LDSX];   // x tile, f16      (18 KB)
  __shared__ _Float16 wt[BN * LDSX];   // dequant W tile   (18 KB)
  __shared__ float    tab[16];         // NF4 table: 16 entries -> 16 distinct
                                       // banks -> conflict-free per-lane gather

  const int tid  = threadIdx.x;
  const int lane = tid & 31;
  const int wave = tid >> 5;
  const int wm   = wave >> 1;   // 0..3 -> M offset wm*32
  const int wn   = wave & 1;    // 0..1 -> N offset wn*64
  const int lrow = lane & 15;
  const int kh   = lane >> 4;   // lane-half selects K sub-range in fragments

  const int blockM = blockIdx.x * BM;
  const int blockN = blockIdx.y * BN;

  if (tid < 16) tab[tid] = NF4_TAB[tid];

  v8f acc[2][4];
#pragma unroll
  for (int i = 0; i < 2; ++i)
#pragma unroll
    for (int j = 0; j < 4; ++j) acc[i][j] = (v8f)(0.0f);

  // staging work partition
  const int xid   = tid * 8;   // 8 x float4 of the x tile per thread
  const int wrow  = tid >> 1;  // qweight row (2 threads per output row)
  const int whalf = tid & 1;   // which 32-wide K half of the slab

  // register staging for the software pipeline
  v4f xreg[8];
  v4i qreg[4];

  auto load_slab = [&](int kb) {
    const int k0 = kb * BK;
#pragma unroll
    for (int i = 0; i < 8; ++i) {
      const int id = xid + i;
      const int row = id >> 4, c4 = id & 15;
      xreg[i] = *(const v4f*)(x + (size_t)(blockM + row) * IN_F + k0 + c4 * 4);
    }
#pragma unroll
    for (int i = 0; i < 4; ++i)
      qreg[i] = *(const v4i*)(qw + (size_t)(blockN + wrow) * (IN_F / 2) +
                              kb * 32 + whalf * 16 + i * 4);
  };

  load_slab(0);
  __syncthreads();  // tab visible

  for (int kb = 0; kb < NSLAB; ++kb) {
    const float s = scales[kb];  // scalar per slab (SGPR broadcast)

    __syncthreads();  // previous slab's WMMA fragment reads done

    // ---- commit staged x tile: 128 rows x 64 k, f32 -> f16 -----------------
#pragma unroll
    for (int i = 0; i < 8; ++i) {
      const int id = xid + i;
      const int row = id >> 4, c4 = id & 15;
      v4h hv = {(_Float16)xreg[i].x, (_Float16)xreg[i].y, (_Float16)xreg[i].z,
                (_Float16)xreg[i].w};
      *(v4h*)(xs + row * LDSX + c4 * 4) = hv;  // ds_store_b64
    }

    // ---- commit staged W tile: dequant NF4 (2 nibbles / byte-as-int32) -----
#pragma unroll
    for (int i = 0; i < 4; ++i) {
      v8h hv;
#pragma unroll
      for (int j = 0; j < 4; ++j) {
        const int b = qreg[i][j] & 0xff;
        hv[2 * j]     = (_Float16)(tab[b & 15] * s);
        hv[2 * j + 1] = (_Float16)(tab[(b >> 4) & 15] * s);
      }
      *(v8h*)(wt + wrow * LDSX + whalf * 32 + i * 8) = hv;  // ds_store_b128
    }

    __syncthreads();

    // ---- issue next slab's global loads; latency hidden under WMMAs --------
    if (kb + 1 < NSLAB) load_slab(kb + 1);
    if (kb + 2 < NSLAB) {  // light touch two slabs out (global_prefetch_b8)
      __builtin_prefetch(
          x + (size_t)(blockM + (xid >> 4)) * IN_F + (kb + 2) * BK, 0, 3);
      __builtin_prefetch(
          qw + (size_t)(blockN + wrow) * (IN_F / 2) + (kb + 2) * 32 +
              whalf * 16, 0, 3);
    }

    // ---- two K=32 WMMA steps per slab --------------------------------------
#pragma unroll
    for (int ks = 0; ks < BK; ks += 32) {
      v16h afrag[2];
#pragma unroll
      for (int mf = 0; mf < 2; ++mf) {
        // A 16-bit 16x32 layout: lane m=lrow, VGPR0-3 <- K = kh*8..+7,
        // VGPR4-7 <- K = 16+kh*8..+7
        const _Float16* p = xs + (wm * 32 + mf * 16 + lrow) * LDSX + ks + kh * 8;
        v8h lo = *(const v8h*)p;          // ds_load_b128
        v8h hi = *(const v8h*)(p + 16);   // ds_load_b128
        afrag[mf] = __builtin_shufflevector(lo, hi, 0, 1, 2, 3, 4, 5, 6, 7, 8,
                                            9, 10, 11, 12, 13, 14, 15);
      }
      v16h bfrag[4];
#pragma unroll
      for (int nf = 0; nf < 4; ++nf) {
        // B 32x16 layout: lane n=lrow; VGPR0-3 <- K = kh*8..+7, VGPR4-7 <- K = 16+kh*8..+7
        const _Float16* p =
            wt + (wn * 64 + nf * 16 + lrow) * LDSX + ks + kh * 8;
        v8h lo = *(const v8h*)p;          // ds_load_b128
        v8h hi = *(const v8h*)(p + 16);   // ds_load_b128
        bfrag[nf] = __builtin_shufflevector(lo, hi, 0, 1, 2, 3, 4, 5, 6, 7, 8,
                                            9, 10, 11, 12, 13, 14, 15);
      }
#pragma unroll
      for (int mf = 0; mf < 2; ++mf)
#pragma unroll
        for (int nf = 0; nf < 4; ++nf)
          acc[mf][nf] = __builtin_amdgcn_wmma_f32_16x16x32_f16(
              false, afrag[mf], false, bfrag[nf], (short)0, acc[mf][nf],
              false, false);
    }
  }

  // ---- epilogue: out = acc + sum_r t[m][r] * B[n][r]  (t pre-scaled by 4) --
  v4f bn0[4], bn1[4];
#pragma unroll
  for (int nf = 0; nf < 4; ++nf) {
    const int gn = blockN + wn * 64 + nf * 16 + lrow;
    bn0[nf] = *(const v4f*)(loraB + (size_t)gn * RANK);
    bn1[nf] = *(const v4f*)(loraB + (size_t)gn * RANK + 4);
  }
#pragma unroll
  for (int mf = 0; mf < 2; ++mf) {
#pragma unroll
    for (int v = 0; v < 8; ++v) {
      // C/D f32 layout: VGPR v holds M = v (lanes 0-15) / v+8 (lanes 16-31)
      const int gm = blockM + wm * 32 + mf * 16 + kh * 8 + v;
      v4f ta = *(const v4f*)(t + (size_t)gm * RANK);
      v4f tb = *(const v4f*)(t + (size_t)gm * RANK + 4);
#pragma unroll
      for (int nf = 0; nf < 4; ++nf) {
        const float l = ta.x * bn0[nf].x + ta.y * bn0[nf].y +
                        ta.z * bn0[nf].z + ta.w * bn0[nf].w +
                        tb.x * bn1[nf].x + tb.y * bn1[nf].y +
                        tb.z * bn1[nf].z + tb.w * bn1[nf].w;
        const int gn = blockN + wn * 64 + nf * 16 + lrow;
        out[(size_t)gm * OUT_F + gn] = acc[mf][nf][v] + l;
      }
    }
  }
}

// ---------------------------------------------------------------------------
// Host-side launch. d_in order: x, qweight, scales, lora_A, lora_B.
// d_ws holds t = 4*(x @ A^T): 8192*8 floats = 256 KB.
// ---------------------------------------------------------------------------
extern "C" void kernel_launch(void* const* d_in, const int* in_sizes, int n_in,
                              void* d_out, int out_size, void* d_ws,
                              size_t ws_size, hipStream_t stream) {
  const float* x      = (const float*)d_in[0];
  const int*   qw     = (const int*)d_in[1];
  const float* scales = (const float*)d_in[2];
  const float* loraA  = (const float*)d_in[3];
  const float* loraB  = (const float*)d_in[4];
  float*       out    = (float*)d_out;
  float*       t      = (float*)d_ws;   // TOKENS*RANK floats

  lora_xa_kernel<<<TOKENS / 256, 256, 0, stream>>>(x, loraA, t);

  dim3 grid(TOKENS / BM, OUT_F / BN);
  qlora_wmma_kernel<<<grid, 256, 0, stream>>>(x, qw, scales, loraB, t, out);
}
